// QuantileCombinedModel_24833500905678
// MI455X (gfx1250) — matrix operand order repaired and checked by reading, not compiled, not run on hardware
//
#include <hip/hip_runtime.h>
#include <hip/hip_bf16.h>

#define Hs   256
#define Isz  99
#define Bsz  64
#define Tsz  128
#define DECs 325
#define Psz  9
#define KX   128            // input size padded 99 -> 128
#define NWG  16             // persistent workgroups (== max cluster size)
#define NTHR 256            // 8 waves per WG (wave32)
#define MROWS (Bsz*Tsz)     // 8192 encoder rows

#define AS1 __attribute__((address_space(1)))
#define AS3 __attribute__((address_space(3)))

#if defined(__has_builtin)
#if __has_builtin(__builtin_amdgcn_global_load_async_to_lds_b128) && \
    __has_builtin(__builtin_amdgcn_s_wait_asynccnt)
#define HAVE_ASYNC 1
#endif
#endif
#ifndef HAVE_ASYNC
#define HAVE_ASYNC 0
#endif

typedef __attribute__((ext_vector_type(16))) _Float16 v16h;
typedef __attribute__((ext_vector_type(8)))  _Float16 v8h;
typedef __attribute__((ext_vector_type(8)))  float    v8f;
typedef int __attribute__((ext_vector_type(4)))       ai4;

// LDS partition (dynamic shared, 192 KB total):
//   [0,112K)    weight slice, fragment-order tiles: Wih0(16) Whh0(32) Wih1(32) Whh1(32)
//   [112K,144K) hlA (h0 staging, 64x256 f16)
//   [144K,176K) hlB (h1 staging)
//   [176K,192K) gl  (gate tiles, 4x64x16 f32)
#define OFF_W    0
#define OFF_HLA  (112*1024)
#define OFF_HLB  (144*1024)
#define OFF_GL   (176*1024)
#define SMEM_BYTES (192*1024)
// tile offsets (in halves; one tile = 32 lanes * 16 halves = 512)
#define TW_IH0   0
#define TW_HH0   (16*512)
#define TW_IH1   (48*512)
#define TW_HH1   (80*512)

// ---------------------------------------------------------------- WMMA core
__device__ __forceinline__ v8f wmma_f16(v16h a, v16h b, v8f c) {
  return __builtin_amdgcn_wmma_f32_16x16x32_f16(
      /*neg_a=*/false, a, /*neg_b=*/false, b,
      /*c_mod=*/(short)0, c, /*reuse_a=*/false, /*reuse_b=*/false);
}

// A fragment: row-major [M x K] source, tile (m0,k0).
__device__ __forceinline__ v16h load_a16(const _Float16* base, int ld, int m0, int k0) {
  int lane = threadIdx.x & 31;
  int r = lane & 15, hi = lane >> 4;
  const _Float16* row = base + (size_t)(m0 + r) * (size_t)ld + k0;
  v8h x0 = *(const v8h*)(row + 16 * hi);
  v8h x1 = *(const v8h*)(row + 16 * hi + 8);
  v16h a;
#pragma unroll
  for (int j = 0; j < 8; ++j) { a[j] = x0[j]; a[8 + j] = x1[j]; }
  return a;
}

__device__ __forceinline__ v16h load_a16_relu(const _Float16* base, int ld, int m0, int k0) {
  v16h a = load_a16(base, ld, m0, k0);
#pragma unroll
  for (int j = 0; j < 16; ++j) a[j] = (a[j] > (_Float16)0.f) ? a[j] : (_Float16)0.f;
  return a;
}

// B fragment from row-major W [N x K] global (heads)
__device__ __forceinline__ v16h load_b16(const _Float16* W, int ld, int n0, int k0) {
  int lane = threadIdx.x & 31;
  int n = lane & 15, kh = (lane >> 4) * 16;
  return *(const v16h*)(W + (size_t)(n0 + n) * (size_t)ld + k0 + kh);
}

// B fragment from LDS in fragment order: tile-major, 32 B per lane, conflict-free
__device__ __forceinline__ v16h load_b_lds(const _Float16* wl, int tile) {
  int lane = threadIdx.x & 31;
  return *(const v16h*)(wl + (((size_t)tile * 32 + lane) << 4));
}

// ---------------------------------------------------------------- utilities
__global__ void zero_u32(unsigned* p, long n) {
  for (long i = blockIdx.x * (long)blockDim.x + threadIdx.x; i < n;
       i += (long)gridDim.x * blockDim.x)
    p[i] = 0u;
}

__global__ void cvt_f16_pad(const float* src, _Float16* dst,
                            int srcRows, int dstRows, int srcK, int dstK) {
  long total = (long)dstRows * dstK;
  for (long i = blockIdx.x * (long)blockDim.x + threadIdx.x; i < total;
       i += (long)gridDim.x * blockDim.x) {
    int r = (int)(i / dstK), k = (int)(i % dstK);
    float v = (r < srcRows && k < srcK) ? src[(size_t)r * srcK + k] : 0.f;
    dst[i] = (_Float16)v;
  }
}

// ---------------------------------------------------------------- persistent LSTM
struct Params {
  const _Float16 *Wih0, *Whh0, *Wih1, *Whh1, *Wlin, *Wp1, *Wp2;
  const float *b0, *b1, *blin, *bp1, *bp2;
  const _Float16 *x16;                 // [B*T, 128] f16 (padded input)
  _Float16 *out0, *out1;               // [B*T, 256] f16
  _Float16 *h0, *h1, *xin, *zdec;      // states
  float *c0, *c1;                      // cell states fp32
  unsigned *bar;                       // grid barrier counters
  float *sf, *yy;                      // d_out regions: sensors_forward, yields
};

// device-scope step barrier (fresh counter per sync point, zeroed each call)
// + cluster barrier (NOP unless dispatched as a 16-WG cluster)
__device__ __forceinline__ void gbar(unsigned* bar, int slot, int tid) {
  __syncthreads();
  if (tid == 0) {
    __hip_atomic_fetch_add(&bar[slot], 1u, __ATOMIC_ACQ_REL, __HIP_MEMORY_SCOPE_AGENT);
    while (__hip_atomic_load(&bar[slot], __ATOMIC_ACQUIRE, __HIP_MEMORY_SCOPE_AGENT) <
           (unsigned)NWG)
      __builtin_amdgcn_s_sleep(2);
  }
  __builtin_amdgcn_s_cluster_barrier();   // s_barrier_signal/-wait -3
  __syncthreads();
  __threadfence();
}

// repack this WG's 16-wide hidden slice of W [4H x K] into fragment-order LDS tiles
__device__ void repack_w(const _Float16* W, int ld, int hs0,
                         _Float16* dst, int ntx, int tid) {
  int lane = tid & 31, wave = tid >> 5;
  int n = lane & 15, kh = (lane >> 4) * 16;
  for (int tI = wave; tI < 4 * ntx; tI += 8) {
    int g = tI / ntx, ki = tI % ntx;
    int n0 = g * Hs + hs0;
    v16h b = *(const v16h*)(W + (size_t)(n0 + n) * ld + ki * 32 + kh);
    *(v16h*)(dst + (((size_t)tI * 32 + lane) << 4)) = b;
  }
}

// One LSTM layer step for this WG's 16-wide hidden slice.
__device__ void layer_step(const _Float16* X, int ldx, int ntx,
                           const _Float16* wihl, const _Float16* whhl,
                           const float* bias,
                           const _Float16* hl, float* gl,
                           float* cg, _Float16* hg,
                           _Float16* outg, int ldout,
                           int wg, int tid) {
  int wave = tid >> 5, lane = tid & 31;
  int hs0 = wg * 16;
  int n = lane & 15, hi = lane >> 4;
  for (int tI = wave; tI < 16; tI += 8) {          // 16 tiles (4 gates x 4 M), 2/wave
    int g = tI >> 2, mi = tI & 3;
    int m0 = mi * 16;
    v8f acc = {};
    for (int ki = 0; ki < ntx; ++ki)               // input projection (B from LDS)
      acc = wmma_f16(load_a16(X, ldx, m0, ki * 32), load_b_lds(wihl, g * ntx + ki), acc);
#pragma unroll
    for (int ki = 0; ki < 8; ++ki)                 // recurrent (A + B from LDS)
      acc = wmma_f16(load_a16(hl, Hs, m0, ki * 32), load_b_lds(whhl, g * 8 + ki), acc);
    float bb = bias[g * Hs + hs0 + n];
#pragma unroll
    for (int v = 0; v < 8; ++v)
      gl[((g * 64) + m0 + hi * 8 + v) * 16 + n] = acc[v] + bb;
  }
  __syncthreads();
  for (int e = tid; e < 64 * 16; e += NTHR) {      // cell update, 64x16 slice
    int m = e >> 4, nn = e & 15;
    int col = hs0 + nn;
    float gi = gl[(0 * 64 + m) * 16 + nn];
    float gf = gl[(1 * 64 + m) * 16 + nn];
    float gc = gl[(2 * 64 + m) * 16 + nn];
    float go = gl[(3 * 64 + m) * 16 + nn];
    float si = 1.f / (1.f + __expf(-gi));
    float sf_ = 1.f / (1.f + __expf(-gf));
    float so = 1.f / (1.f + __expf(-go));
    float c = sf_ * cg[m * Hs + col] + si * tanhf(gc);
    float h = so * tanhf(c);
    cg[m * Hs + col] = c;
    _Float16 hh = (_Float16)h;
    hg[m * Hs + col] = hh;
    if (outg) outg[(size_t)m * ldout + col] = hh;
  }
}

// heads phase 1: sens = relu(h1)@Wlin^T + blin  (-> sensors_forward, next xin)
//                z    = relu(relu(h1)@Wp1^T + bp1)
__device__ void heads_sens_z(const Params& p, int t, int wg, int tid, bool full) {
  int wave = tid >> 5, lane = tid & 31;
  int gw = wg * 8 + wave;
  int n = lane & 15, hi = lane >> 4;
  if (gw < 28) {                          // sens: 4 M x 7 N tiles (N padded 112)
    int mi = gw / 7, ni = gw % 7;
    int m0 = mi * 16, n0 = ni * 16;
    v8f acc = {};
#pragma unroll
    for (int k = 0; k < Hs; k += 32)
      acc = wmma_f16(load_a16_relu(p.h1, Hs, m0, k), load_b16(p.Wlin, Hs, n0, k), acc);
    int col = n0 + n;
    float bb = (col < Isz) ? p.blin[col] : 0.f;
#pragma unroll
    for (int v = 0; v < 8; ++v) {
      int m = m0 + hi * 8 + v;
      float val = acc[v] + bb;
      if (col < Isz) {
        if (t >= 0) p.sf[((size_t)m * DECs + t) * Isz + col] = val;
        p.xin[m * KX + col] = (_Float16)val;
      } else {
        p.xin[m * KX + col] = (_Float16)0.f;
      }
    }
  } else if (full && gw < 60) {           // z: 4 M x 8 N tiles
    int idx = gw - 28;
    int mi = idx / 8, ni = idx % 8;
    int m0 = mi * 16, n0 = ni * 16;
    v8f acc = {};
#pragma unroll
    for (int k = 0; k < Hs; k += 32)
      acc = wmma_f16(load_a16_relu(p.h1, Hs, m0, k), load_b16(p.Wp1, Hs, n0, k), acc);
    int col = n0 + n;
    float bb = p.bp1[col];
#pragma unroll
    for (int v = 0; v < 8; ++v) {
      int m = m0 + hi * 8 + v;
      float zz = acc[v] + bb;
      zz = zz > 0.f ? zz : 0.f;
      p.zdec[m * KX + col] = (_Float16)zz;
    }
  }
}

// heads phase 2: yld = z @ Wp2^T + bp2  (N padded 9->16); runs on 4 waves
__device__ void heads_yld(const Params& p, int t, int wg, int tid) {
  int wave = tid >> 5, lane = tid & 31;
  int gw = wg * 8 + wave;
  if (gw >= 4) return;
  int m0 = gw * 16;
  int n = lane & 15, hi = lane >> 4;
  v8f acc = {};
#pragma unroll
  for (int k = 0; k < KX; k += 32)
    acc = wmma_f16(load_a16(p.zdec, KX, m0, k), load_b16(p.Wp2, KX, 0, k), acc);
  if (n < Psz) {
    float bb = p.bp2[n];
#pragma unroll
    for (int v = 0; v < 8; ++v) {
      int m = m0 + hi * 8 + v;
      p.yy[((size_t)m * (Tsz + DECs) + Tsz + t) * Psz + n] = acc[v] + bb;
    }
  }
}

__global__ __launch_bounds__(NTHR) void lstm_persistent(Params p) {
  extern __shared__ char smem[];
  _Float16* wsl = (_Float16*)(smem + OFF_W);
  _Float16* hlA = (_Float16*)(smem + OFF_HLA);
  _Float16* hlB = (_Float16*)(smem + OFF_HLB);
  float*    gl  = (float*)(smem + OFF_GL);
  int tid = threadIdx.x;
  int wg = blockIdx.x;
  int slot = 0;

  // Stage one 64x256 f16 state matrix into LDS at byte offset `offB`.
  auto stage = [&](int offB, const _Float16* src) {
#if HAVE_ASYNC
    char* d = smem + offB;
    for (int i = tid; i < 2048; i += NTHR)          // 2048 x 16B
      __builtin_amdgcn_global_load_async_to_lds_b128(
          (AS1 ai4*)(const_cast<_Float16*>(src) + i * 8),
          (AS3 ai4*)(d + i * 16), 0, 0);
#else
    _Float16* d = (_Float16*)(smem + offB);
    for (int i = tid; i < 2048; i += NTHR)
      ((v8h*)d)[i] = ((const v8h*)src)[i];
#endif
  };
  auto stage_wait = [&]() {
#if HAVE_ASYNC
    __builtin_amdgcn_s_wait_asynccnt(0);
#endif
    __syncthreads();
  };

  // Warm L2 with the padded weights (global_prefetch)
  {
    const char* w0 = (const char*)p.Whh0;
    const char* w1 = (const char*)p.Whh1;
    const char* w2 = (const char*)p.Wih1;
    for (size_t off = (size_t)tid * 128; off < (size_t)1024 * Hs * 2;
         off += (size_t)NTHR * 128) {
      __builtin_prefetch(w0 + off, 0, 3);
      __builtin_prefetch(w1 + off, 0, 3);
      __builtin_prefetch(w2 + off, 0, 3);
    }
  }

  // Pin this WG's recurrent weight slices in LDS (fragment order), once.
  repack_w(p.Wih0, KX, wg * 16, wsl + TW_IH0, 4, tid);
  repack_w(p.Whh0, Hs, wg * 16, wsl + TW_HH0, 8, tid);
  repack_w(p.Wih1, Hs, wg * 16, wsl + TW_IH1, 8, tid);
  repack_w(p.Whh1, Hs, wg * 16, wsl + TW_HH1, 8, tid);
  __syncthreads();

  // ---- encoder layer 0
  for (int t = 0; t < Tsz; ++t) {
    stage(OFF_HLA, p.h0);
    stage_wait();
    layer_step(p.x16 + (size_t)t * KX, Tsz * KX, 4, wsl + TW_IH0, wsl + TW_HH0, p.b0,
               hlA, gl, p.c0, p.h0, p.out0 + (size_t)t * Hs, Tsz * Hs, wg, tid);
    gbar(p.bar, slot++, tid);
  }
  // ---- encoder layer 1
  for (int t = 0; t < Tsz; ++t) {
    stage(OFF_HLB, p.h1);
    stage_wait();
    layer_step(p.out0 + (size_t)t * Hs, Tsz * Hs, 8, wsl + TW_IH1, wsl + TW_HH1, p.b1,
               hlB, gl, p.c1, p.h1, p.out1 + (size_t)t * Hs, Tsz * Hs, wg, tid);
    gbar(p.bar, slot++, tid);
  }
  // ---- decoder seed: xin = lin_head(h1_final)
  heads_sens_z(p, -1, wg, tid, false);
  gbar(p.bar, slot++, tid);
  // ---- decoder: 325 closed-loop steps, 3 grid barriers each
  for (int t = 0; t < DECs; ++t) {
    stage(OFF_HLA, p.h0);      // needed now
    stage(OFF_HLB, p.h1);      // overlaps layer-0 compute
#if HAVE_ASYNC
    __builtin_amdgcn_s_wait_asynccnt(8);   // first 8 per-thread loads (hlA) done
#endif
    __syncthreads();
    layer_step(p.xin, KX, 4, wsl + TW_IH0, wsl + TW_HH0, p.b0,
               hlA, gl, p.c0, p.h0, nullptr, 0, wg, tid);
    if (t > 0) heads_yld(p, t - 1, wg, tid);       // folded into this phase
    gbar(p.bar, slot++, tid);
    stage_wait();                                  // hlB ready
    layer_step(hlA, Hs, 8, wsl + TW_IH1, wsl + TW_HH1, p.b1,
               hlB, gl, p.c1, p.h1, nullptr, 0, wg, tid);
    gbar(p.bar, slot++, tid);
    heads_sens_z(p, t, wg, tid, true);
    gbar(p.bar, slot++, tid);
  }
  heads_yld(p, DECs - 1, wg, tid);
}

// ---------------------------------------------------------------- encoder heads (parallel)
__global__ __launch_bounds__(NTHR) void enc_heads1(const _Float16* out1, const _Float16* Wlin,
                                                   const _Float16* Wp1, const float* blin,
                                                   const float* bp1, float* sb, _Float16* zbig) {
  int wave = threadIdx.x >> 5, lane = threadIdx.x & 31;
  int gw = blockIdx.x * 8 + wave;
  int n = lane & 15, hi = lane >> 4;
  const int MT = MROWS / 16;              // 512 M-tiles
  if (gw < MT * 7) {                      // sensors_back
    int mi = gw / 7, ni = gw % 7;
    int m0 = mi * 16, n0 = ni * 16;
    v8f acc = {};
#pragma unroll
    for (int k = 0; k < Hs; k += 32)
      acc = wmma_f16(load_a16_relu(out1, Hs, m0, k), load_b16(Wlin, Hs, n0, k), acc);
    int col = n0 + n;
    if (col < Isz) {
      float bb = blin[col];
#pragma unroll
      for (int v = 0; v < 8; ++v) {
        int r = m0 + hi * 8 + v;          // r = b*T + t
        sb[(size_t)r * Isz + col] = acc[v] + bb;
      }
    }
  } else if (gw < MT * 15) {              // z = relu(relu(out1)@Wp1^T + bp1)
    int idx = gw - MT * 7;
    int mi = idx / 8, ni = idx % 8;
    int m0 = mi * 16, n0 = ni * 16;
    v8f acc = {};
#pragma unroll
    for (int k = 0; k < Hs; k += 32)
      acc = wmma_f16(load_a16_relu(out1, Hs, m0, k), load_b16(Wp1, Hs, n0, k), acc);
    int col = n0 + n;
    float bb = bp1[col];
#pragma unroll
    for (int v = 0; v < 8; ++v) {
      int r = m0 + hi * 8 + v;
      float zz = acc[v] + bb;
      zz = zz > 0.f ? zz : 0.f;
      zbig[(size_t)r * KX + col] = (_Float16)zz;
    }
  }
}

__global__ __launch_bounds__(NTHR) void enc_heads2(const _Float16* zbig, const _Float16* Wp2,
                                                   const float* bp2, float* yy) {
  int wave = threadIdx.x >> 5, lane = threadIdx.x & 31;
  int gw = blockIdx.x * 8 + wave;
  if (gw >= MROWS / 16) return;
  int n = lane & 15, hi = lane >> 4;
  int m0 = gw * 16;
  v8f acc = {};
#pragma unroll
  for (int k = 0; k < KX; k += 32)
    acc = wmma_f16(load_a16(zbig, KX, m0, k), load_b16(Wp2, KX, 0, k), acc);
  if (n < Psz) {
    float bb = bp2[n];
#pragma unroll
    for (int v = 0; v < 8; ++v) {
      int r = m0 + hi * 8 + v;            // r = b*T + t
      int b = r >> 7, tt = r & 127;
      yy[((size_t)b * (Tsz + DECs) + tt) * Psz + n] = acc[v] + bb;
    }
  }
}

// ---------------------------------------------------------------- launch
extern "C" void kernel_launch(void* const* d_in, const int* in_sizes, int n_in,
                              void* d_out, int out_size, void* d_ws, size_t ws_size,
                              hipStream_t stream) {
  (void)in_sizes; (void)n_in; (void)out_size; (void)ws_size;
  const float* x    = (const float*)d_in[0];
  const float* Wih0 = (const float*)d_in[1];
  const float* Whh0 = (const float*)d_in[2];
  const float* b0   = (const float*)d_in[3];
  const float* Wih1 = (const float*)d_in[4];
  const float* Whh1 = (const float*)d_in[5];
  const float* b1   = (const float*)d_in[6];
  const float* Wlin = (const float*)d_in[7];
  const float* blin = (const float*)d_in[8];
  const float* Wp1  = (const float*)d_in[9];
  const float* bp1  = (const float*)d_in[10];
  const float* Wp2  = (const float*)d_in[11];
  const float* bp2  = (const float*)d_in[12];

  char* ws = (char*)d_ws;
  size_t off = 0;
  auto take = [&](size_t bytes) {
    size_t r = off;
    off = (off + bytes + 255) & ~(size_t)255;
    return r;
  };
  size_t oWih0 = take((size_t)1024 * 128 * 2);
  size_t oWhh0 = take((size_t)1024 * 256 * 2);
  size_t oWih1 = take((size_t)1024 * 256 * 2);
  size_t oWhh1 = take((size_t)1024 * 256 * 2);
  size_t oWlin = take((size_t)112 * 256 * 2);
  size_t oWp1  = take((size_t)128 * 256 * 2);
  size_t oWp2  = take((size_t)16 * 128 * 2);
  size_t oX16  = take((size_t)MROWS * 128 * 2);
  size_t oOut0 = take((size_t)MROWS * 256 * 2);
  size_t oOut1 = take((size_t)MROWS * 256 * 2);
  size_t oZbig = take((size_t)MROWS * 128 * 2);
  size_t zstart = off;                      // ---- zero-initialized state region
  size_t oH0   = take((size_t)64 * 256 * 2);
  size_t oH1   = take((size_t)64 * 256 * 2);
  size_t oXin  = take((size_t)64 * 128 * 2);
  size_t oZdec = take((size_t)64 * 128 * 2);
  size_t oC0   = take((size_t)64 * 256 * 4);
  size_t oC1   = take((size_t)64 * 256 * 4);
  size_t oBar  = take((size_t)2048 * 4);
  size_t zbytes = off - zstart;

  float* out = (float*)d_out;
  float* sf = out;                                        // [B, DEC, I]
  float* sb = out + (size_t)Bsz * DECs * Isz;             // [B, T, I]
  float* yy = sb + (size_t)Bsz * Tsz * Isz;               // [B, T+DEC, P]

  zero_u32<<<64, NTHR, 0, stream>>>((unsigned*)(ws + zstart), (long)(zbytes / 4));

  cvt_f16_pad<<<256, NTHR, 0, stream>>>(Wih0, (_Float16*)(ws + oWih0), 1024, 1024, 99, 128);
  cvt_f16_pad<<<256, NTHR, 0, stream>>>(Whh0, (_Float16*)(ws + oWhh0), 1024, 1024, 256, 256);
  cvt_f16_pad<<<256, NTHR, 0, stream>>>(Wih1, (_Float16*)(ws + oWih1), 1024, 1024, 256, 256);
  cvt_f16_pad<<<256, NTHR, 0, stream>>>(Whh1, (_Float16*)(ws + oWhh1), 1024, 1024, 256, 256);
  cvt_f16_pad<<<64,  NTHR, 0, stream>>>(Wlin, (_Float16*)(ws + oWlin), 99, 112, 256, 256);
  cvt_f16_pad<<<64,  NTHR, 0, stream>>>(Wp1,  (_Float16*)(ws + oWp1),  128, 128, 256, 256);
  cvt_f16_pad<<<8,   NTHR, 0, stream>>>(Wp2,  (_Float16*)(ws + oWp2),  9, 16, 128, 128);
  cvt_f16_pad<<<512, NTHR, 0, stream>>>(x,    (_Float16*)(ws + oX16),  MROWS, MROWS, 99, 128);

  Params P;
  P.Wih0 = (const _Float16*)(ws + oWih0);
  P.Whh0 = (const _Float16*)(ws + oWhh0);
  P.Wih1 = (const _Float16*)(ws + oWih1);
  P.Whh1 = (const _Float16*)(ws + oWhh1);
  P.Wlin = (const _Float16*)(ws + oWlin);
  P.Wp1  = (const _Float16*)(ws + oWp1);
  P.Wp2  = (const _Float16*)(ws + oWp2);
  P.b0 = b0; P.b1 = b1; P.blin = blin; P.bp1 = bp1; P.bp2 = bp2;
  P.x16  = (const _Float16*)(ws + oX16);
  P.out0 = (_Float16*)(ws + oOut0);
  P.out1 = (_Float16*)(ws + oOut1);
  P.h0   = (_Float16*)(ws + oH0);
  P.h1   = (_Float16*)(ws + oH1);
  P.xin  = (_Float16*)(ws + oXin);
  P.zdec = (_Float16*)(ws + oZdec);
  P.c0   = (float*)(ws + oC0);
  P.c1   = (float*)(ws + oC1);
  P.bar  = (unsigned*)(ws + oBar);
  P.sf = sf; P.yy = yy;

  lstm_persistent<<<NWG, NTHR, SMEM_BYTES, stream>>>(P);

  enc_heads1<<<(MROWS / 16) * 15 / 8, NTHR, 0, stream>>>(
      P.out1, P.Wlin, P.Wp1, blin, bp1, sb, (_Float16*)(ws + oZbig));
  enc_heads2<<<(MROWS / 16) / 8, NTHR, 0, stream>>>(
      (const _Float16*)(ws + oZbig), P.Wp2, bp2, yy);
}
